// SimilarityLoss_13778255086094
// MI455X (gfx1250) — matrix operand order repaired and checked by reading, not compiled, not run on hardware
//
#include <hip/hip_runtime.h>

// ---------------------------------------------------------------------------
// SimilarityLoss fused pipeline for MI455X (gfx1250, wave32, WMMA bf16x3).
//
//   map_1 = apply_cc(y, y')  : per-batch GEMM  patches(y')[400x1024] x im2col(y)[1024x5929]
//   map_2 = apply_cc(y', y)  : roles swapped
//   out   = mean over (b,pixel) of cosine(map_1[:,px], map_2[:,px])  (P=400 axis)
//
// Stage 1 (prep_split)   : fp32 -> bf16 hi/lo NCHW copies of y and y' (for B/im2col)
// Stage 2 (prep_swizzleA): patch matrices pre-swizzled into this file's wave32 WMMA
//                          A-fragment layout (k = 16*half + e), hi/lo -> b128 loads in GEMM
// Stage 3 (simloss_main) : fused double-GEMM + cosine, 5 waves x 5 M-tiles (static,
//                          no exec masking), bf16x3 accumulation
// Stage 4 (finalize)     : deterministic tree reduction of per-workgroup partials
//
// d_ws layout (bytes):
//   [0, 256)                 : reserved
//   base + k*A, k=0..7       : bf16 arrays of NELEM elems:
//       0:y_hi 1:y_lo 2:yp_hi 3:yp_lo 4:Ay_hi 5:Ay_lo 6:Ayp_hi 7:Ayp_lo
//   base + 8*A               : float parts[BATCH*NJT]
//   total ~52.4 MB (entire working set is L2-resident on the 192MB L2)
// ---------------------------------------------------------------------------

typedef __bf16 bf16x16 __attribute__((ext_vector_type(16)));
typedef __bf16 bf16x4  __attribute__((ext_vector_type(4)));
typedef float  f32x8   __attribute__((ext_vector_type(8)));
typedef float  f32x4   __attribute__((ext_vector_type(4)));

#define BATCH 8
#define CH    64
#define HH    80
#define WW    80
#define PW_   20                    // patches per spatial dim
#define HO    77
#define WO    77
#define NCOL  (HO * WO)             // 5929
#define NJT   ((NCOL + 15) / 16)    // 371 column tiles
#define KDIM  (CH * 16)             // 1024
#define NKB   (KDIM / 32)           // 32 K-blocks (2 channels each)
#define MTILES 25                   // 400 / 16
#define NWAVE 5                     // 5 waves x 5 M-tiles = 25 (statically uniform)
#define TPW   5                     // M-tiles per wave
#define NELEM ((size_t)BATCH * CH * HH * WW)   // 3,276,800 per tensor
#define NPART (BATCH * NJT)         // 2968 per-workgroup partials
#define OUT_SCALE 0.001f
#define EPS_ 1e-8f

// ---------------------------------------------------------------------------
// Stage 1: fp32 -> bf16 hi/lo split, vectorized x4, both tensors.
// hi = bf16(v), lo = bf16(v - float(hi)).
// ---------------------------------------------------------------------------
__global__ void __launch_bounds__(256)
prep_split(const float* __restrict__ y, const float* __restrict__ yp,
           __bf16* __restrict__ wsb) {
  size_t tid = (size_t)blockIdx.x * 256 + threadIdx.x;   // < 2*NELEM/4
  int t = (tid >= NELEM / 4) ? 1 : 0;
  size_t i4 = tid - (size_t)t * (NELEM / 4);
  const float* src = t ? yp : y;
  __bf16* hi = wsb + (size_t)(2 * t) * NELEM;
  __bf16* lo = hi + NELEM;

  f32x4 v = *(const f32x4*)(src + i4 * 4);
  bf16x4 h, l;
#pragma unroll
  for (int k = 0; k < 4; ++k) {
    __bf16 hv = (__bf16)v[k];
    h[k] = hv;
    l[k] = (__bf16)(v[k] - (float)hv);
  }
  *(bf16x4*)(hi + i4 * 4) = h;
  *(bf16x4*)(lo + i4 * 4) = l;
}

// ---------------------------------------------------------------------------
// Stage 2: build A-fragments in wave32 WMMA layout.
// A 16x32 bf16 fragment (contiguous k order used here): lane = half*16 + m, element e:
//   M = m,  K = e + half*16
// One thread per (tensor, batch, mt, kb, lane); 16 hi + 16 lo bf16 contiguous.
// ---------------------------------------------------------------------------
__global__ void __launch_bounds__(256)
prep_swizzleA(const float* __restrict__ y, const float* __restrict__ yp,
              __bf16* __restrict__ wsb) {
  int tid = blockIdx.x * 256 + threadIdx.x;   // < 2*8*25*32*32 = 409600
  int lane = tid & 31;  int r = tid >> 5;
  int kb = r & 31;      r >>= 5;
  int mt = r % MTILES;  r /= MTILES;
  int b = r & 7;
  int t = r >> 3;                              // 0: from y, 1: from y'
  const float* src = (t ? yp : y) + (size_t)b * CH * HH * WW;

  size_t frag = ((((size_t)b * MTILES + mt) * NKB + kb) * 32 + lane) * 16;
  __bf16* hi = wsb + (size_t)(4 + 2 * t) * NELEM + frag;
  __bf16* lo = hi + NELEM;

  const int half = lane >> 4;
  const int m = lane & 15;
  const int p = mt * 16 + m;                   // patch id < 400
  const int pr = p / PW_, pc = p - pr * PW_;
  const int ry = 4 * pr, rx = 4 * pc;
  const int c0 = 2 * kb;

  bf16x16 H, L;
#pragma unroll
  for (int e = 0; e < 16; ++e) {
    const int K = e + (half << 4);
    const int c = c0 + (K >> 4);
    const int kk = K & 15;
    const float v = src[((size_t)c * HH + (ry + (kk >> 2))) * WW + (rx + (kk & 3))];
    __bf16 hv = (__bf16)v;
    H[e] = hv;
    L[e] = (__bf16)(v - (float)hv);
  }
  *(bf16x16*)hi = H;
  *(bf16x16*)lo = L;
}

// B-fragment (32x16 bf16): lane = half*16 + n, element e:  N = n, K = e + 16*half
//   -> c = c0 + half, (di,dj) = (e>>2, e&3).  `extra` (0 or NELEM) selects hi/lo;
//   all displacements are compile-time -> fold into load immediate offsets.
__device__ __forceinline__ bf16x16 load_b_frag(const __bf16* __restrict__ base, size_t extra) {
  bf16x16 f;
#pragma unroll
  for (int e = 0; e < 16; ++e) f[e] = base[extra + (size_t)(e >> 2) * WW + (e & 3)];
  return f;
}

#define WMMA_BF16(A, Bm, Cacc) \
  __builtin_amdgcn_wmma_f32_16x16x32_bf16(false, (A), false, (Bm), (short)0, (Cacc), false, false)

// ---------------------------------------------------------------------------
// Stage 3: fused double GEMM + cosine.  One workgroup = (batch, 16 pixels),
// 160 threads = 5 waves; wave w statically owns M-tiles {5w .. 5w+4} for BOTH
// maps -> no divergence, no exec masking, no accumulator copies.
// Per wave per kb: 2 base-pointer setups, 40 b128 A-loads + 64 u16 B-loads
// (all immediate-offset), 30 WMMAs in two independent accumulation chains.
// ---------------------------------------------------------------------------
__global__ void __launch_bounds__(160)
simloss_main(const __bf16* __restrict__ wsb, float* __restrict__ parts) {
  const int jt = blockIdx.x;
  const int b  = blockIdx.y;
  const int tid = threadIdx.x;
  const int wave = tid >> 5;                   // 0..4
  const int lane = tid & 31;
  const int half = lane >> 4;
  const int ncol = lane & 15;

  const int j0 = jt * 16;
  int j = j0 + ncol;
  if (j >= NCOL) j = NCOL - 1;                 // clamp loads; masked in reduce
  const int oy = j / WO, ox = j - oy * WO;

  const __bf16* yh  = wsb;                     // im2col(y)  hi (lo at +NELEM)
  const __bf16* yph = wsb + 2 * NELEM;         // im2col(y') hi (lo at +NELEM)
  const __bf16* ayh = wsb + 4 * NELEM;         // patches(y)  hi -> A of map2
  const __bf16* aph = wsb + 6 * NELEM;         // patches(y') hi -> A of map1

  f32x8 acc1[TPW], acc2[TPW];
#pragma unroll
  for (int i = 0; i < TPW; ++i) { acc1[i] = {}; acc2[i] = {}; }

  const size_t bimg = (size_t)b * CH * HH * WW;
  const int mt0 = wave * TPW;                  // first M-tile of this wave

  for (int kb = 0; kb < NKB; ++kb) {
    const int c0 = 2 * kb;
    // B fragments: hi at +0, lo at +NELEM (6.55MB -> fits 24-bit imm offset).
    const size_t bb = bimg + ((size_t)(c0 + half) * HH + oy) * WW + ox;
    const bf16x16 b1h = load_b_frag(yh + bb,  0);
    const bf16x16 b1l = load_b_frag(yh + bb,  NELEM);
    const bf16x16 b2h = load_b_frag(yph + bb, 0);
    const bf16x16 b2l = load_b_frag(yph + bb, NELEM);

    // A fragments: one base pointer per tensor; M-tile step = NKB*512 elems
    // (32KB) and hi/lo step = NELEM (6.55MB) are compile-time immediates.
    const size_t ao = ((((size_t)b * MTILES + mt0) * NKB + kb) * 32 + lane) * 16;
    const __bf16* pA1 = aph + ao;              // map1 A = patches(y')
    const __bf16* pA2 = ayh + ao;              // map2 A = patches(y)
#pragma unroll
    for (int i = 0; i < TPW; ++i) {
      const size_t mo = (size_t)i * (NKB * 512);
      const bf16x16 a1h = *(const bf16x16*)(pA1 + mo);
      const bf16x16 a1l = *(const bf16x16*)(pA1 + mo + NELEM);
      const bf16x16 a2h = *(const bf16x16*)(pA2 + mo);
      const bf16x16 a2l = *(const bf16x16*)(pA2 + mo + NELEM);
      // bf16x3: ah*bh + ah*bl + al*bh (al*bl ~2^-16, dropped); two
      // independent accumulation chains interleaved to keep the XDL pipe full.
      acc1[i] = WMMA_BF16(a1h, b1h, acc1[i]);
      acc2[i] = WMMA_BF16(a2h, b2h, acc2[i]);
      acc1[i] = WMMA_BF16(a1h, b1l, acc1[i]);
      acc2[i] = WMMA_BF16(a2h, b2l, acc2[i]);
      acc1[i] = WMMA_BF16(a1l, b1h, acc1[i]);
      acc2[i] = WMMA_BF16(a2l, b2h, acc2[i]);
    }
  }

  // D layout: VGPR r, lane -> M = mt*16 + r + 8*half, N = ncol.
  float num = 0.f, s1 = 0.f, s2 = 0.f;
#pragma unroll
  for (int i = 0; i < TPW; ++i) {
#pragma unroll
    for (int r = 0; r < 8; ++r) {
      const float m1 = acc1[i][r] * OUT_SCALE;
      const float m2 = acc2[i][r] * OUT_SCALE;
      num = fmaf(m1, m2, num);
      s1  = fmaf(m1, m1, s1);
      s2  = fmaf(m2, m2, s2);
    }
  }

  // Deterministic reduction: lane-pair fold (same pixel column), per-wave LDS
  // slots, fixed-order cross-wave sum, per-workgroup partial to global.
  num += __shfl_xor(num, 16, 32);
  s1  += __shfl_xor(s1, 16, 32);
  s2  += __shfl_xor(s2, 16, 32);

  __shared__ float red[NWAVE][3][16];
  __shared__ float cosv[16];
  if (lane < 16) {
    red[wave][0][lane] = num;
    red[wave][1][lane] = s1;
    red[wave][2][lane] = s2;
  }
  __syncthreads();
  if (tid < 16) {
    float n = 0.f, a = 0.f, c = 0.f;
#pragma unroll
    for (int w = 0; w < NWAVE; ++w) {
      n += red[w][0][tid]; a += red[w][1][tid]; c += red[w][2][tid];
    }
    float cv = 0.f;
    if (j0 + tid < NCOL) {
      const float d1 = fmaxf(sqrtf(a), EPS_);
      const float d2 = fmaxf(sqrtf(c), EPS_);
      cv = n / (d1 * d2);
    }
    cosv[tid] = cv;
  }
  __syncthreads();
  if (tid == 0) {
    float s = 0.f;
#pragma unroll
    for (int k = 0; k < 16; ++k) s += cosv[k];
    parts[b * NJT + jt] = s;
  }
}

// Stage 4: deterministic tree reduction of the 2968 partials -> mean.
__global__ void __launch_bounds__(256)
simloss_finalize(const float* __restrict__ parts, float* __restrict__ out) {
  __shared__ float sh[256];
  float s = 0.f;
  for (int i = threadIdx.x; i < NPART; i += 256) s += parts[i];
  sh[threadIdx.x] = s;
  __syncthreads();
  for (int off = 128; off > 0; off >>= 1) {
    if ((int)threadIdx.x < off) sh[threadIdx.x] += sh[threadIdx.x + off];
    __syncthreads();
  }
  if (threadIdx.x == 0) out[0] = sh[0] * (1.0f / (float)(BATCH * NCOL));
}

extern "C" void kernel_launch(void* const* d_in, const int* in_sizes, int n_in,
                              void* d_out, int out_size, void* d_ws, size_t ws_size,
                              hipStream_t stream) {
  const float* y  = (const float*)d_in[0];   // (8,64,80,80) fp32
  const float* yp = (const float*)d_in[1];   // (8,64,80,80) fp32
  float* out = (float*)d_out;                // scalar fp32

  __bf16* wsb  = (__bf16*)((char*)d_ws + 256);
  float* parts = (float*)((char*)d_ws + 256 + 8 * NELEM * sizeof(__bf16));

  // Stage 1: bf16 hi/lo NCHW copies of both tensors (2*NELEM/4 threads).
  prep_split<<<(unsigned)((2 * NELEM / 4) / 256), 256, 0, stream>>>(y, yp, wsb);
  // Stage 2: WMMA-swizzled patch (A) matrices, hi/lo, both tensors.
  prep_swizzleA<<<(2 * BATCH * MTILES * NKB * 32) / 256, 256, 0, stream>>>(y, yp, wsb);
  // Stage 3: fused double GEMM + cosine per (batch, 16-pixel tile).
  dim3 grid(NJT, BATCH);
  simloss_main<<<grid, 160, 0, stream>>>(wsb, parts);
  // Stage 4: deterministic mean.
  simloss_finalize<<<1, 256, 0, stream>>>(parts, out);
}
